// MultiHeadedAttention_34084860461307
// MI455X (gfx1250) — matrix operand order repaired and checked by reading, not compiled, not run on hardware
//
#include <hip/hip_runtime.h>
#include <hip/hip_bf16.h>

typedef __attribute__((ext_vector_type(16))) _Float16 v16h;
typedef __attribute__((ext_vector_type(8)))  _Float16 v8h;
typedef __attribute__((ext_vector_type(4)))  _Float16 v4h;
typedef __attribute__((ext_vector_type(8)))  float    v8f;
typedef __attribute__((ext_vector_type(4)))  float    v4f;

#define DIM_B 2
#define DIM_S 2048
#define DIM_D 1024
#define DIM_H 16
#define DIM_DK 64

__device__ __forceinline__ v8f wmma_f16(v16h a, v16h b, v8f c) {
  return __builtin_amdgcn_wmma_f32_16x16x32_f16(
      /*neg_a=*/false, a, /*neg_b=*/false, b,
      /*c_mod=*/(short)0, c, /*reuse_a=*/false, /*reuse_b=*/false);
}

// ---- A-fragment (16x32 f16, M x K): lane holds row M=lane&15.
// halves h: K = (h<8?0:16) + 8*(lane>=16) + (h&7)  -> two contiguous groups of 8.
__device__ __forceinline__ v16h loadA(const _Float16* __restrict__ base,
                                      int row0, int ld, int k0, int lane) {
  int nsel = lane & 15;
  int kg   = (lane >> 4) << 3;           // 0 or 8
  const _Float16* p = base + (size_t)(row0 + nsel) * ld + k0 + kg;
  v8h lo = *(const v8h*)p;
  v8h hi = *(const v8h*)(p + 16);
  v16h r;
#pragma unroll
  for (int i = 0; i < 8; ++i) { r[i] = lo[i]; r[i + 8] = hi[i]; }
  return r;
}

// ---- B-fragment (32x16 f16, K x N): lane holds col N=lane&15.
// halves h: K = (h<8?0:16) + 8*(lane>=16) + (h&7)  -> two contiguous groups of 8.
__device__ __forceinline__ v16h loadB(const _Float16* __restrict__ base,
                                      int col0, int ld, int k0, int lane) {
  int nsel = lane & 15;
  int kg   = (lane >> 4) << 3;           // 0 or 8
  const _Float16* p = base + (size_t)(col0 + nsel) * ld + k0 + kg;
  v8h lo = *(const v8h*)p;
  v8h hi = *(const v8h*)(p + 16);
  v16h r;
#pragma unroll
  for (int i = 0; i < 8; ++i) { r[i] = lo[i]; r[i + 8] = hi[i]; }
  return r;
}

// ---------------- f32 -> f16 bulk convert (vectorized, bandwidth bound)
__global__ __launch_bounds__(256) void cvt_kernel(const v4f* __restrict__ src,
                                                  v4h* __restrict__ dst, int n4) {
  int i = blockIdx.x * 256 + threadIdx.x;
  if (i < n4) {
    v4f x = src[i];
    v4h y;
#pragma unroll
    for (int j = 0; j < 4; ++j) y[j] = (_Float16)x[j];
    dst[i] = y;
  }
}

// ---------------- Projection GEMM: Y = X @ W^T + bias, one wave -> 32x64 tile.
// 8 WMMAs per k-step against 12 b128 loads (2x A-reuse, all-f16 operands).
// MODE 0: f16 out as [B,H,S,DK]   (Q with scale=1/8, K with scale=1)
// MODE 1: f16 out as [B,H,DK,S]   (V^T)
// MODE 2: f32 out as [B*S, D]     (final projection to d_out)
template <int MODE>
__global__ __launch_bounds__(128) void proj_kernel(
    const _Float16* __restrict__ X, const _Float16* __restrict__ W,
    const float* __restrict__ bias, _Float16* __restrict__ outh,
    float* __restrict__ outf, float scale) {
  const int D = DIM_D;
  int lane = threadIdx.x & 31;
  int w    = (blockIdx.x << 2) + (threadIdx.x >> 5);   // 4 waves / block
  int tn = w & 15;            // 16 tiles of 64 columns
  int tm = w >> 4;            // 128 tiles of 32 rows
  int m0 = tm << 5, n0 = tn << 6;

  v8f acc0[4] = {}, acc1[4] = {};
  for (int k0 = 0; k0 < D; k0 += 32) {
    v16h a0 = loadA(X, m0,      D, k0, lane);
    v16h a1 = loadA(X, m0 + 16, D, k0, lane);
#pragma unroll
    for (int d = 0; d < 4; ++d) {
      v16h b = loadB(W, n0 + (d << 4), D, k0, lane);   // B(k,n)=W[n,k]
      acc0[d] = wmma_f16(a0, b, acc0[d]);
      acc1[d] = wmma_f16(a1, b, acc1[d]);
    }
  }

  int halfid = lane >> 4, nsel = lane & 15;
#pragma unroll
  for (int g = 0; g < 2; ++g) {
#pragma unroll
    for (int d = 0; d < 4; ++d) {
      int ncol = n0 + (d << 4) + nsel;
      float bv = bias[ncol];
#pragma unroll
      for (int r = 0; r < 8; ++r) {
        int mrow = m0 + (g << 4) + r + (halfid << 3);
        float y = ((g ? acc1[d][r] : acc0[d][r]) + bv) * scale;
        if constexpr (MODE == 0) {
          int bb = mrow >> 11, s = mrow & (DIM_S - 1);
          int hh = ncol >> 6, dk = ncol & (DIM_DK - 1);
          outh[((((size_t)bb << 4) + hh) * DIM_S + s) * DIM_DK + dk] = (_Float16)y;
        } else if constexpr (MODE == 1) {
          int bb = mrow >> 11, s = mrow & (DIM_S - 1);
          int hh = ncol >> 6, dk = ncol & (DIM_DK - 1);
          outh[((((size_t)bb << 4) + hh) * DIM_DK + dk) * DIM_S + s] = (_Float16)y;
        } else {
          outf[(size_t)mrow * DIM_D + ncol] = y;
        }
      }
    }
  }
}

// ---------------- Flash attention: one wave per (b, h, 16-row query tile).
__global__ __launch_bounds__(128) void attn_kernel(
    const _Float16* __restrict__ Qh, const _Float16* __restrict__ Kh,
    const _Float16* __restrict__ Vt, _Float16* __restrict__ Ah) {
  __shared__ __align__(16) _Float16 smem[4][16][40];   // stride 40 -> no bank conflicts
  const int S = DIM_S, DK = DIM_DK, H = DIM_H;
  int lane = threadIdx.x & 31;
  int wiw  = threadIdx.x >> 5;
  int w    = (blockIdx.x << 2) + wiw;                  // 0..4095
  int st = w & 127;
  int hh = (w >> 7) & 15;
  int b  = w >> 11;
  int s0 = st << 4;
  const _Float16* Qb = Qh + ((size_t)b * H + hh) * (size_t)S * DK;
  const _Float16* Kb = Kh + ((size_t)b * H + hh) * (size_t)S * DK;
  const _Float16* Vb = Vt + ((size_t)b * H + hh) * (size_t)DK * S;
  int halfid = lane >> 4, nsel = lane & 15;

  // Q fragments for the whole tile (Q pre-scaled by 1/sqrt(DK))
  v16h qa0 = loadA(Qb, s0, DK, 0, lane);
  v16h qa1 = loadA(Qb, s0, DK, 32, lane);

  float mrow[8], lrow[8];
  v8f acc[4] = {};
#pragma unroll
  for (int r = 0; r < 8; ++r) { mrow[r] = -3.0e38f; lrow[r] = 0.0f; }

  int nkb = (s0 >> 5) + 1;                 // causal: key blocks of 32, t0 <= s0
  for (int kb = 0; kb < nkb; ++kb) {
    int t0 = kb << 5;
    // scores: two 16x16 tiles over key cols [t0, t0+32)
    v8f sc0 = {}, sc1 = {};
    {
      v16h k00 = loadB(Kb, t0, DK, 0, lane);
      v16h k01 = loadB(Kb, t0, DK, 32, lane);
      sc0 = wmma_f16(qa0, k00, sc0);
      sc0 = wmma_f16(qa1, k01, sc0);
      v16h k10 = loadB(Kb, t0 + 16, DK, 0, lane);
      v16h k11 = loadB(Kb, t0 + 16, DK, 32, lane);
      sc1 = wmma_f16(qa0, k10, sc1);
      sc1 = wmma_f16(qa1, k11, sc1);
    }
    if (t0 + 31 > s0) {                    // diagonal block: causal mask
#pragma unroll
      for (int r = 0; r < 8; ++r) {
        int s = s0 + r + (halfid << 3);
        if (t0 + nsel > s)      sc0[r] = -1.0e30f;
        if (t0 + 16 + nsel > s) sc1[r] = -1.0e30f;
      }
    }
    // online softmax; rows live across the 16-lane half-group
    float rm[8];
#pragma unroll
    for (int r = 0; r < 8; ++r) rm[r] = fmaxf(sc0[r], sc1[r]);
#pragma unroll
    for (int off = 1; off < 16; off <<= 1)
#pragma unroll
      for (int r = 0; r < 8; ++r) rm[r] = fmaxf(rm[r], __shfl_xor(rm[r], off, 32));

    float p0[8], p1[8], rs[8];
#pragma unroll
    for (int r = 0; r < 8; ++r) {
      float mn = fmaxf(mrow[r], rm[r]);
      float cs = __expf(mrow[r] - mn);
      mrow[r] = mn;
      p0[r] = __expf(sc0[r] - mn);
      p1[r] = __expf(sc1[r] - mn);
      rs[r] = p0[r] + p1[r];
      lrow[r] *= cs;
#pragma unroll
      for (int d = 0; d < 4; ++d) acc[d][r] *= cs;
    }
#pragma unroll
    for (int off = 1; off < 16; off <<= 1)
#pragma unroll
      for (int r = 0; r < 8; ++r) rs[r] += __shfl_xor(rs[r], off, 32);
#pragma unroll
    for (int r = 0; r < 8; ++r) lrow[r] += rs[r];

    // transpose P (C-layout -> A-layout) through LDS, as f16
#pragma unroll
    for (int r = 0; r < 8; ++r) {
      int row = r + (halfid << 3);
      smem[wiw][row][nsel]      = (_Float16)p0[r];
      smem[wiw][row][16 + nsel] = (_Float16)p1[r];
    }
    v16h pa;
    {
      int kg = halfid << 3;
      const _Float16* lp = &smem[wiw][nsel][kg];
      v8h lo = *(const v8h*)lp;
      v8h hi = *(const v8h*)(lp + 16);
#pragma unroll
      for (int i = 0; i < 8; ++i) { pa[i] = lo[i]; pa[i + 8] = hi[i]; }
    }
    // P(16x32) @ V(32x64): B(k=t,n=dk) = Vt[dk*S + t0 + k], contiguous in k
#pragma unroll
    for (int d = 0; d < 4; ++d) {
      v16h vb = loadB(Vb, d << 4, S, t0, lane);
      acc[d] = wmma_f16(pa, vb, acc[d]);
    }
  }

  // epilogue: normalize and store attention output [B*S, D] f16
#pragma unroll
  for (int d = 0; d < 4; ++d) {
#pragma unroll
    for (int r = 0; r < 8; ++r) {
      float y = acc[d][r] / lrow[r];
      int s = s0 + r + (halfid << 3);
      int col = (hh << 6) + (d << 4) + nsel;
      Ah[((size_t)b * S + s) * DIM_D + col] = (_Float16)y;
    }
  }
}

extern "C" void kernel_launch(void* const* d_in, const int* in_sizes, int n_in,
                              void* d_out, int out_size, void* d_ws, size_t ws_size,
                              hipStream_t stream) {
  (void)in_sizes; (void)n_in; (void)out_size; (void)ws_size;
  const float* q  = (const float*)d_in[0];
  const float* k  = (const float*)d_in[1];
  const float* v  = (const float*)d_in[2];
  // d_in[3] = mask: causal, computed analytically -> never read
  const float* Wq = (const float*)d_in[4];
  const float* bq = (const float*)d_in[5];
  const float* Wk = (const float*)d_in[6];
  const float* bk = (const float*)d_in[7];
  const float* Wv = (const float*)d_in[8];
  const float* bv = (const float*)d_in[9];
  const float* Wo = (const float*)d_in[10];
  const float* bo = (const float*)d_in[11];
  float* out = (float*)d_out;

  const size_t N  = (size_t)DIM_B * DIM_S * DIM_D;   // 4M elements
  const size_t NW = (size_t)DIM_D * DIM_D;           // 1M elements
  _Float16* Qh  = (_Float16*)d_ws;          // [B,H,S,DK]
  _Float16* Kh  = Qh + N;                   // [B,H,S,DK]
  _Float16* Vt  = Qh + 2 * N;               // [B,H,DK,S]
  _Float16* Xq  = Qh + 3 * N;               // f16 query input; reused as Ah
  _Float16* Xk  = Qh + 4 * N;
  _Float16* Xv  = Qh + 5 * N;
  _Float16* Wqh = Qh + 6 * N;
  _Float16* Wkh = Wqh + NW;
  _Float16* Wvh = Wqh + 2 * NW;
  _Float16* Woh = Wqh + 3 * NW;
  _Float16* Ah  = Xq;                       // alias: Xq dead after Q projection

  dim3 blk128(128);
  dim3 gridProj(512);    // 2048 waves, 32x64 tiles
  dim3 gridAttn(1024);   // 4096 waves

  // Bulk f32 -> f16 conversions (one-time; removes all conversion VALU from GEMMs)
  int n4  = (int)(N / 4);
  int nw4 = (int)(NW / 4);
  cvt_kernel<<<dim3(n4 / 256), dim3(256), 0, stream>>>((const v4f*)q,  (v4h*)Xq,  n4);
  cvt_kernel<<<dim3(n4 / 256), dim3(256), 0, stream>>>((const v4f*)k,  (v4h*)Xk,  n4);
  cvt_kernel<<<dim3(n4 / 256), dim3(256), 0, stream>>>((const v4f*)v,  (v4h*)Xv,  n4);
  cvt_kernel<<<dim3(nw4 / 256), dim3(256), 0, stream>>>((const v4f*)Wq, (v4h*)Wqh, nw4);
  cvt_kernel<<<dim3(nw4 / 256), dim3(256), 0, stream>>>((const v4f*)Wk, (v4h*)Wkh, nw4);
  cvt_kernel<<<dim3(nw4 / 256), dim3(256), 0, stream>>>((const v4f*)Wv, (v4h*)Wvh, nw4);
  cvt_kernel<<<dim3(nw4 / 256), dim3(256), 0, stream>>>((const v4f*)Wo, (v4h*)Woh, nw4);

  const float inv_sqrt_dk = 0.125f;   // 1/sqrt(64), folded into Q projection
  proj_kernel<0><<<gridProj, blk128, 0, stream>>>(Xq, Wqh, bq, Qh, nullptr, inv_sqrt_dk);
  proj_kernel<0><<<gridProj, blk128, 0, stream>>>(Xk, Wkh, bk, Kh, nullptr, 1.0f);
  proj_kernel<1><<<gridProj, blk128, 0, stream>>>(Xv, Wvh, bv, Vt, nullptr, 1.0f);
  attn_kernel<<<gridAttn, blk128, 0, stream>>>(Qh, Kh, Vt, Ah);
  proj_kernel<2><<<gridProj, blk128, 0, stream>>>(Ah, Woh, bo, nullptr, out, 1.0f);
}
